// newsModel_18433999635117
// MI455X (gfx1250) — matrix operand order repaired and checked by reading, not compiled, not run on hardware
//
#include <hip/hip_runtime.h>
#include <hip/hip_bf16.h>
#include <math.h>

// ---------------------------------------------------------------------------
// GAT + MLP head for MI455X (gfx1250, wave32).
// Heavy GEMM h = x@W runs on V_WMMA_F32_16X16X32_BF16 (f32 accumulate),
// software-pipelined: tile kt+1 prefetched (non-temporal) while tile kt's
// 8 WMMAs execute against pre-loaded B fragments.
// Edge softmax/aggregation is L2-resident f32 atomics (out = 51MB < 192MB L2).
// ---------------------------------------------------------------------------

typedef __attribute__((ext_vector_type(8)))  float  v8f;
typedef __attribute__((ext_vector_type(4)))  float  v4f;
typedef __attribute__((ext_vector_type(16))) __bf16 v16bf;

union Frag {
  v16bf v;
  unsigned int u[8];
};

__device__ __forceinline__ unsigned short f32_to_bf16(float f) {
  unsigned int u = __builtin_bit_cast(unsigned int, f);
  unsigned int r = u + 0x7FFFu + ((u >> 16) & 1u);   // round-to-nearest-even
  return (unsigned short)(r >> 16);
}

__device__ __forceinline__ float leaky02(float e) {
  return e > 0.0f ? e : 0.2f * e;
}

// ---------------------------------------------------------------------------
// f32 -> bf16 bit conversion (for W / W0, done once; 196 KB each -> L2)
// ---------------------------------------------------------------------------
__global__ void cvt_f32_bf16_kernel(const float* __restrict__ src,
                                    unsigned short* __restrict__ dst, int n) {
  int i = blockIdx.x * blockDim.x + threadIdx.x;
  if (i < n) dst[i] = f32_to_bf16(src[i]);
}

// ---------------------------------------------------------------------------
// WMMA GEMM: Hout[rows x 128] = A(rows x K, f32, optionally row-gathered) @ Wb
// Wb is bf16 bits, row-major [K][128].
// 256 threads = 8 waves; block computes 128 rows x 128 cols.
// Optional epilogue: (a) bias + relu   (b) aS/aD = H . att_{src,dst}
// ---------------------------------------------------------------------------
__global__ __launch_bounds__(256)
void gemm_bf16_wmma_kernel(const float* __restrict__ X, int ldx,
                           const unsigned short* __restrict__ Wb,
                           int K, int Nrows,
                           const int* __restrict__ rowIdx,
                           const float* __restrict__ bias, int fuseBiasRelu,
                           const float* __restrict__ attS,
                           const float* __restrict__ attD,
                           float* __restrict__ Hout,
                           float* __restrict__ aS, float* __restrict__ aD) {
  __shared__ unsigned short ldsA[128][34];   // 32 K + pad (bank-spread)

  const int tid  = threadIdx.x;
  const int lane = tid & 31;
  const int wave = tid >> 5;                 // 0..7 -> 16-row slice
  const int rowBase = blockIdx.x * 128;

  v8f acc[8];
#pragma unroll
  for (int i = 0; i < 8; ++i) acc[i] = (v8f)0.0f;

  // ---- staging assignment: thread loads float4 chunk q of rows r0+32p ----
  const int q  = tid & 7;                    // float4 chunk within 32-wide row
  const int r0 = tid >> 3;                   // 0..31
  const float* srcBase[4];
  bool rowOk[4];
#pragma unroll
  for (int p = 0; p < 4; ++p) {
    const int grow = rowBase + r0 + 32 * p;
    rowOk[p] = (grow < Nrows);
    const int srow = rowOk[p] ? (rowIdx ? rowIdx[grow] : grow) : 0;
    srcBase[p] = X + (size_t)srow * ldx + q * 4;
  }

  v4f stage[4];
#pragma unroll
  for (int p = 0; p < 4; ++p) {
    stage[p] = (v4f)0.0f;
    if (rowOk[p])                            // non-temporal: x is single-pass
      stage[p] = __builtin_nontemporal_load((const v4f*)srcBase[p]);
  }

  const int nkt = K >> 5;                    // K / 32
  for (int kt = 0; kt < nkt; ++kt) {
    // ---- commit staged tile kt to LDS as bf16 ----------------------------
#pragma unroll
    for (int p = 0; p < 4; ++p) {
      const int r  = r0 + 32 * p;
      const int kc = q * 4;
      ldsA[r][kc + 0] = f32_to_bf16(stage[p].x);
      ldsA[r][kc + 1] = f32_to_bf16(stage[p].y);
      ldsA[r][kc + 2] = f32_to_bf16(stage[p].z);
      ldsA[r][kc + 3] = f32_to_bf16(stage[p].w);
    }
    __syncthreads();

    // ---- prefetch tile kt+1 while this tile computes ---------------------
    if (kt + 1 < nkt) {
#pragma unroll
      for (int p = 0; p < 4; ++p) {
        if (rowOk[p])
          stage[p] = __builtin_nontemporal_load(
              (const v4f*)(srcBase[p] + (size_t)(kt + 1) * 32));
      }
    }

    // ---- A fragment (16x32 bf16, ISA layout: lane halves cover K blocks) -
    Frag afr;
    {
      const int arow = wave * 16 + (lane & 15);
      const int kh   = (lane >> 4) * 8;
#pragma unroll
      for (int i = 0; i < 8; ++i) {
        const int k = ((i < 4) ? 0 : 16) + kh + 2 * (i & 3);
        const unsigned int lo = ldsA[arow][k];
        const unsigned int hi = ldsA[arow][k + 1];
        afr.u[i] = lo | (hi << 16);
      }
    }

    // ---- load ALL 8 B fragments first (one clause), then 8 WMMAs ---------
    const int kb = kt * 32 + 8 * (lane >> 4);
    const unsigned short* wrow = Wb + (size_t)kb * 128 + (lane & 15);
    Frag bfr[8];
#pragma unroll
    for (int nt = 0; nt < 8; ++nt) {
#pragma unroll
      for (int i = 0; i < 8; ++i) {
        const int k = ((i < 4) ? 0 : 16) + 2 * (i & 3);
        const unsigned int lo = wrow[(size_t)k * 128 + nt * 16];
        const unsigned int hi = wrow[(size_t)(k + 1) * 128 + nt * 16];
        bfr[nt].u[i] = lo | (hi << 16);
      }
    }
#pragma unroll
    for (int nt = 0; nt < 8; ++nt) {
      acc[nt] = __builtin_amdgcn_wmma_f32_16x16x32_bf16(
          false, afr.v, false, bfr[nt].v, (short)0, acc[nt], false, false);
    }
    __syncthreads();
  }

  // ---- epilogue: store H (C/D layout: vgpr r -> M=r+8*(lane/16), N=lane%16)
  const int m8   = (lane >> 4) * 8;
  const int ncol = lane & 15;
#pragma unroll
  for (int nt = 0; nt < 8; ++nt) {
#pragma unroll
    for (int r = 0; r < 8; ++r) {
      const int grow = rowBase + wave * 16 + m8 + r;
      if (grow < Nrows) {
        float v = acc[nt][r];
        if (fuseBiasRelu) { v += bias[nt * 16 + ncol]; v = fmaxf(v, 0.0f); }
        Hout[(size_t)grow * 128 + nt * 16 + ncol] = v;
      }
    }
  }

  // ---- fused attention dots: aS = H.att_src, aD = H.att_dst --------------
  if (aS != nullptr) {
    float ps[8], pd[8];
#pragma unroll
    for (int r = 0; r < 8; ++r) { ps[r] = 0.0f; pd[r] = 0.0f; }
#pragma unroll
    for (int nt = 0; nt < 8; ++nt) {
      const float s = attS[nt * 16 + ncol];
      const float d = attD[nt * 16 + ncol];
#pragma unroll
      for (int r = 0; r < 8; ++r) {
        ps[r] += acc[nt][r] * s;
        pd[r] += acc[nt][r] * d;
      }
    }
#pragma unroll
    for (int r = 0; r < 8; ++r) {
#pragma unroll
      for (int off = 1; off < 16; off <<= 1) {   // reduce across the N lanes
        ps[r] += __shfl_xor(ps[r], off, 32);
        pd[r] += __shfl_xor(pd[r], off, 32);
      }
    }
    if ((lane & 15) == 0) {
#pragma unroll
      for (int r = 0; r < 8; ++r) {
        const int grow = rowBase + wave * 16 + m8 + r;
        if (grow < Nrows) { aS[grow] = ps[r]; aD[grow] = pd[r]; }
      }
    }
  }
}

// ---------------------------------------------------------------------------
// GAT softmax + aggregation kernels (self-loop handled as node-level terms)
// ---------------------------------------------------------------------------
__global__ void node_self_score_kernel(const float* __restrict__ aS,
                                       const float* __restrict__ aD,
                                       float* __restrict__ m, int N) {
  int n = blockIdx.x * blockDim.x + threadIdx.x;
  if (n < N) m[n] = leaky02(aS[n] + aD[n]);        // max init = self edge
}

__global__ void edge_score_max_kernel(const int* __restrict__ ei, int E,
                                      const float* __restrict__ aS,
                                      const float* __restrict__ aD,
                                      float* __restrict__ escore,
                                      float* __restrict__ m) {
  int e = blockIdx.x * blockDim.x + threadIdx.x;
  if (e < E) {
    const int s = ei[e];
    const int d = ei[E + e];
    const float v = leaky02(aS[s] + aD[d]);
    escore[e] = v;
    atomicMax(&m[d], v);
  }
}

__global__ void node_denom_init_kernel(const float* __restrict__ aS,
                                       const float* __restrict__ aD,
                                       const float* __restrict__ m,
                                       float* __restrict__ denom,
                                       float* __restrict__ selfexp, int N) {
  int n = blockIdx.x * blockDim.x + threadIdx.x;
  if (n < N) {
    const float ex = expf(leaky02(aS[n] + aD[n]) - m[n]);
    selfexp[n] = ex;
    denom[n]   = ex;
  }
}

__global__ void edge_exp_kernel(const int* __restrict__ ei, int E,
                                const float* __restrict__ m,
                                float* __restrict__ escore,   // in: score, out: exp
                                float* __restrict__ denom) {
  int e = blockIdx.x * blockDim.x + threadIdx.x;
  if (e < E) {
    const int d = ei[E + e];
    const float ex = expf(escore[e] - m[d]);
    escore[e] = ex;
    atomicAdd(&denom[d], ex);
  }
}

__global__ void node_agg_init_kernel(const float* __restrict__ h,
                                     const float* __restrict__ selfexp,
                                     const float* __restrict__ denom,
                                     const float* __restrict__ bconv,
                                     float* __restrict__ out, int N) {
  int i = blockIdx.x * blockDim.x + threadIdx.x;
  if (i < N * 128) {
    const int n = i >> 7;
    const int c = i & 127;
    const float alpha = selfexp[n] / (denom[n] + 1e-16f);
    out[i] = bconv[c] + alpha * h[i];
  }
}

__global__ __launch_bounds__(256)
void edge_aggregate_kernel(const int* __restrict__ ei, int E,
                           const float* __restrict__ h,
                           const float* __restrict__ expe,
                           const float* __restrict__ denom,
                           float* __restrict__ out) {
  const int wid  = (int)((blockIdx.x * blockDim.x + threadIdx.x) >> 5);
  const int lane = threadIdx.x & 31;
  if (wid >= E) return;
  const int s = ei[wid];
  const int d = ei[E + wid];
  const float alpha = expe[wid] / (denom[d] + 1e-16f);
  const float* hs = h   + (size_t)s * 128;
  float*       od = out + (size_t)d * 128;
#pragma unroll
  for (int j = 0; j < 4; ++j) {
    const int c = lane + 32 * j;
    atomicAdd(&od[c], alpha * hs[c]);     // L2-resident (out = 51MB < 192MB)
  }
}

// ---------------------------------------------------------------------------
// Pool / roots / MLP head
// ---------------------------------------------------------------------------
__global__ void zero_kernel(float* __restrict__ p, int n) {
  int i = blockIdx.x * blockDim.x + threadIdx.x;
  if (i < n) p[i] = 0.0f;
}

__global__ void pool_max_kernel(const float* __restrict__ out,
                                const int* __restrict__ batch,
                                float* __restrict__ x1, int N) {
  int i = blockIdx.x * blockDim.x + threadIdx.x;
  if (i < N * 128) {
    const int n = i >> 7;
    const int c = i & 127;
    const float v = fmaxf(out[i], 0.0f);         // relu then segment-max (>=0)
    atomicMax(&x1[batch[n] * 128 + c], v);
  }
}

__global__ void find_roots_kernel(const int* __restrict__ batch, int N, int G,
                                  int* __restrict__ roots) {
  int g = blockIdx.x * blockDim.x + threadIdx.x;
  if (g >= G) return;
  int lo = 0, hi = N;
  while (lo < hi) {                               // searchsorted 'left'
    const int mid = (lo + hi) >> 1;
    if (batch[mid] < g) lo = mid + 1; else hi = mid;
  }
  roots[g] = lo;
}

__global__ __launch_bounds__(128)
void mlp1_kernel(const float* __restrict__ x1, const float* __restrict__ news,
                 const float* __restrict__ W1, const float* __restrict__ b1,
                 float* __restrict__ z2) {
  const int g = blockIdx.x;
  const int c = threadIdx.x;
  float acc = b1[c];
  const float* xg = x1   + g * 128;
  const float* ng = news + g * 128;
  for (int k = 0; k < 128; ++k) acc += xg[k] * W1[k * 128 + c];
  for (int k = 0; k < 128; ++k) acc += ng[k] * W1[(128 + k) * 128 + c];
  z2[g * 128 + c] = fmaxf(acc, 0.0f);
}

__global__ void mlp2_kernel(const float* __restrict__ z2,
                            const float* __restrict__ W2,
                            const float* __restrict__ b2,
                            float* __restrict__ outA, float* __restrict__ outB,
                            int G) {
  int g = blockIdx.x * blockDim.x + threadIdx.x;
  if (g >= G) return;
  const float* z = z2 + g * 128;
  float l0 = b2[0], l1 = b2[1];
  for (int k = 0; k < 128; ++k) {
    l0 += z[k] * W2[k * 2 + 0];
    l1 += z[k] * W2[k * 2 + 1];
  }
  const float mx  = fmaxf(l0, l1);
  const float lse = mx + logf(expf(l0 - mx) + expf(l1 - mx));
  outA[g * 2 + 0] = l0 - lse;
  outA[g * 2 + 1] = l1 - lse;
  outB[g * 2 + 0] = l0 - lse;
  outB[g * 2 + 1] = l1 - lse;
}

// ---------------------------------------------------------------------------
// Host orchestration
// ---------------------------------------------------------------------------
static inline size_t alignUp(size_t x) { return (x + 255) & ~(size_t)255; }

extern "C" void kernel_launch(void* const* d_in, const int* in_sizes, int n_in,
                              void* d_out, int out_size, void* d_ws, size_t ws_size,
                              hipStream_t stream) {
  const float* x        = (const float*)d_in[0];
  const int*   ei       = (const int*)  d_in[1];
  const int*   batch    = (const int*)  d_in[2];
  const float* W        = (const float*)d_in[4];
  const float* att_src  = (const float*)d_in[5];
  const float* att_dst  = (const float*)d_in[6];
  const float* b_conv   = (const float*)d_in[7];
  const float* W0       = (const float*)d_in[8];
  const float* b0       = (const float*)d_in[9];
  const float* W1       = (const float*)d_in[10];
  const float* b1       = (const float*)d_in[11];
  const float* W2       = (const float*)d_in[12];
  const float* b2       = (const float*)d_in[13];

  const int N = in_sizes[2];            // 100000 nodes
  const int E = in_sizes[1] / 2;        // 1.6M edges
  const int F = in_sizes[0] / N;        // 768 (K for both GEMMs)
  const int H = in_sizes[5];            // 128 (kernels assume 128)
  const int G = out_size / (H + 4);     // 128 graphs (logp|x1|logp layout)

  // workspace carve-up
  char* ws = (char*)d_ws;
  size_t off = 0;
  float* h        = (float*)(ws + off); off += alignUp((size_t)N * H * 4);
  float* outconv  = (float*)(ws + off); off += alignUp((size_t)N * H * 4);
  float* aS       = (float*)(ws + off); off += alignUp((size_t)N * 4);
  float* aD       = (float*)(ws + off); off += alignUp((size_t)N * 4);
  float* mmax     = (float*)(ws + off); off += alignUp((size_t)N * 4);
  float* denom    = (float*)(ws + off); off += alignUp((size_t)N * 4);
  float* selfexp  = (float*)(ws + off); off += alignUp((size_t)N * 4);
  float* escore   = (float*)(ws + off); off += alignUp((size_t)E * 4);
  unsigned short* Wb  = (unsigned short*)(ws + off); off += alignUp((size_t)F * H * 2);
  unsigned short* W0b = (unsigned short*)(ws + off); off += alignUp((size_t)F * H * 2);
  int*   roots    = (int*)  (ws + off); off += alignUp((size_t)G * 4);
  float* news     = (float*)(ws + off); off += alignUp((size_t)G * H * 4);
  float* z2       = (float*)(ws + off); off += alignUp((size_t)G * H * 4);
  (void)ws_size; (void)n_in;

  float* logpA = (float*)d_out;                 // [G,2]
  float* x1    = (float*)d_out + 2 * G;         // [G,H]
  float* logpB = (float*)d_out + 2 * G + G * H; // [G,2]

  // 1) weights -> bf16 (once per launch; tiny, L2-resident afterwards)
  {
    const int n = F * H;
    cvt_f32_bf16_kernel<<<(n + 255) / 256, 256, 0, stream>>>(W,  Wb,  n);
    cvt_f32_bf16_kernel<<<(n + 255) / 256, 256, 0, stream>>>(W0, W0b, n);
  }

  // 2) h = x @ W  (WMMA) + fused attention dots
  {
    const int blocks = (N + 127) / 128;
    gemm_bf16_wmma_kernel<<<blocks, 256, 0, stream>>>(
        x, F, Wb, F, N, nullptr, nullptr, 0, att_src, att_dst, h, aS, aD);
  }

  // 3) segment softmax over edges (+ self loops) grouped by dst
  node_self_score_kernel<<<(N + 255) / 256, 256, 0, stream>>>(aS, aD, mmax, N);
  edge_score_max_kernel<<<(E + 255) / 256, 256, 0, stream>>>(ei, E, aS, aD, escore, mmax);
  node_denom_init_kernel<<<(N + 255) / 256, 256, 0, stream>>>(aS, aD, mmax, denom, selfexp, N);
  edge_exp_kernel<<<(E + 255) / 256, 256, 0, stream>>>(ei, E, mmax, escore, denom);

  // 4) aggregation: out = b_conv + sum_e alpha * h[src]
  node_agg_init_kernel<<<((N * 128) + 255) / 256, 256, 0, stream>>>(
      h, selfexp, denom, b_conv, outconv, N);
  {
    const long long waves = E;
    const long long blocks = (waves * 32 + 255) / 256;
    edge_aggregate_kernel<<<(int)blocks, 256, 0, stream>>>(ei, E, h, escore, denom, outconv);
  }

  // 5) relu + per-graph max pool -> x1 (written directly into d_out slice)
  zero_kernel<<<(out_size + 255) / 256, 256, 0, stream>>>((float*)d_out, out_size);
  pool_max_kernel<<<((N * 128) + 255) / 256, 256, 0, stream>>>(outconv, batch, x1, N);

  // 6) roots + news = relu(x[roots] @ W0 + b0)  (WMMA, gathered rows)
  find_roots_kernel<<<(G + 127) / 128, 128, 0, stream>>>(batch, N, G, roots);
  gemm_bf16_wmma_kernel<<<(G + 127) / 128, 256, 0, stream>>>(
      x, F, W0b, F, G, roots, b0, 1, nullptr, nullptr, news, nullptr, nullptr);

  // 7) MLP head + log_softmax (outputs: logp, x1, logp)
  mlp1_kernel<<<G, 128, 0, stream>>>(x1, news, W1, b1, z2);
  mlp2_kernel<<<(G + 127) / 128, 128, 0, stream>>>(z2, W2, b2, logpA, logpB, G);
}
